// SPVBlock_8469675508142
// MI455X (gfx1250) — matrix operand order repaired and checked by reading, not compiled, not run on hardware
//
#include <hip/hip_runtime.h>

// ---------------- types ----------------
typedef __bf16  bf16x16 __attribute__((ext_vector_type(16)));
typedef float   f32x8   __attribute__((ext_vector_type(8)));
typedef unsigned short u16;
typedef unsigned int   u32;

union BFrag { bf16x16 v; uint4 q[2]; u16 h[16]; };

// ---------------- problem constants ----------------
#define NVOX   250000
#define NPART  125000
#define NPTS   600000
#define NNEXT  100000
#define NDS    100000
#define CC     64
#define KK     27
#define SXC    240
#define SYC    180
#define SZC    16
#define NCELL  1382400           // 2*240*180*16
#define OUTFEA 38400000          // NPTS*64

// ---------------- helpers ----------------
__device__ __forceinline__ u16 f2bf(float f) {
  union { float f; u32 u; } c; c.f = f;
  u32 u = c.u;
  u32 r = u + 0x7FFFu + ((u >> 16) & 1u);
  return (u16)(r >> 16);
}
__device__ __forceinline__ float leakyf(float v) { return v >= 0.f ? v : 0.1f * v; }

__device__ __forceinline__ f32x8 wmma_bf16(bf16x16 a, bf16x16 b, f32x8 c) {
  return __builtin_amdgcn_wmma_f32_16x16x32_bf16(false, a, false, b, (short)0, c, false, false);
}
// A fragment: 16x32 bf16 tile. base = row*Kd + kc*32 + half*8 (elements)
__device__ __forceinline__ bf16x16 load_a(const u16* X, int base) {
  const uint4* p = reinterpret_cast<const uint4*>(X + base);
  BFrag f; f.q[0] = p[0]; f.q[1] = p[2];
  return f.v;
}
// B fragment: pre-swizzled, fully contiguous per lane
__device__ __forceinline__ bf16x16 load_b(const u16* wf, int kc, int NT, int t, int lane) {
  const uint4* p = reinterpret_cast<const uint4*>(wf + (((kc * NT + t) * 32 + lane) << 4));
  BFrag f; f.q[0] = p[0]; f.q[1] = p[1];
  return f.v;
}

// ---------------- utility kernels ----------------
__global__ __launch_bounds__(256) void k_zero(float* p, int n) {
  int i = blockIdx.x * 256 + threadIdx.x;
  if (i < n) p[i] = 0.f;
}
__global__ __launch_bounds__(256) void k_cvt_bf16(const float* s, u16* d, int n) {
  int i = blockIdx.x * 256 + threadIdx.x;
  if (i < n) d[i] = f2bf(s[i]);
}
// W logical [Ktot, Nd] row-major f32 -> fragment-ordered bf16
__global__ __launch_bounds__(256) void k_swz(const float* W, u16* dst, int Ktot, int Nd, int total) {
  int idx = blockIdx.x * 256 + threadIdx.x;
  if (idx >= total) return;
  int e    = idx & 15;
  int lane = (idx >> 4) & 31;
  int rem  = idx >> 9;
  int NT   = Nd >> 4;
  int t    = rem % NT;
  int kc   = rem / NT;
  int kabs = kc * 32 + (lane >> 4) * 8 + (e & 7) + ((e >> 3) << 4);
  int nabs = t * 16 + (lane & 15);
  dst[idx] = f2bf(W[kabs * Nd + nabs]);
}

// ---------------- gathered submanifold conv: out = sum_k x[nbr[n,k]] @ W[k] ----------------
__global__ __launch_bounds__(32) void k_gconv(const u16* X, const int* nbr, const u16* wf,
                                              float* outF, int Nr) {
  int lane = threadIdx.x;
  int mrow = lane & 15, half = lane >> 4;
  int n0 = blockIdx.x * 16;
  int ic = n0 + mrow; if (ic >= Nr) ic = Nr - 1;
  f32x8 z8 = {0.f,0.f,0.f,0.f,0.f,0.f,0.f,0.f};
  f32x8 acc[4]; acc[0]=z8; acc[1]=z8; acc[2]=z8; acc[3]=z8;
  for (int k = 0; k < KK; ++k) {
    int row = nbr[ic * KK + k];
    #pragma unroll
    for (int cc = 0; cc < 2; ++cc) {
      bf16x16 a = load_a(X, row * 64 + cc * 32 + half * 8);
      int kc = k * 2 + cc;
      #pragma unroll
      for (int t = 0; t < 4; ++t) {
        bf16x16 b = load_b(wf, kc, 4, t, lane);
        acc[t] = wmma_bf16(a, b, acc[t]);
      }
    }
  }
  #pragma unroll
  for (int t = 0; t < 4; ++t)
    #pragma unroll
    for (int r = 0; r < 8; ++r) {
      int gi = n0 + half * 8 + r;
      if (gi < Nr) outF[gi * 64 + t * 16 + mrow] = acc[t][r];
    }
}

// ---------------- dense GEMM (opt gather rows, bias, leaky, f32/bf16 out) ----------------
__global__ __launch_bounds__(32) void k_dgemm(const u16* X, const int* rowidx, const u16* wf,
                                              const float* bias, float* outF, u16* outB,
                                              int obStride, int obOff,
                                              int Nr, int Kd, int Nd, int doLeaky) {
  int lane = threadIdx.x;
  int mrow = lane & 15, half = lane >> 4;
  int n0 = blockIdx.x * 16;
  int ic = n0 + mrow; if (ic >= Nr) ic = Nr - 1;
  int row = rowidx ? rowidx[ic] : ic;
  int KC = Kd >> 5, NT = Nd >> 4;
  f32x8 z8 = {0.f,0.f,0.f,0.f,0.f,0.f,0.f,0.f};
  f32x8 acc[4]; acc[0]=z8; acc[1]=z8; acc[2]=z8; acc[3]=z8;
  for (int kc = 0; kc < KC; ++kc) {
    bf16x16 a = load_a(X, row * Kd + kc * 32 + half * 8);
    for (int t = 0; t < NT; ++t) {
      bf16x16 b = load_b(wf, kc, NT, t, lane);
      acc[t] = wmma_bf16(a, b, acc[t]);
    }
  }
  for (int t = 0; t < NT; ++t)
    #pragma unroll
    for (int r = 0; r < 8; ++r) {
      int gi = n0 + half * 8 + r;
      if (gi < Nr) {
        float v = acc[t][r];
        if (bias) v += bias[t * 16 + mrow];
        if (doLeaky) v = leakyf(v);
        if (outF) outF[gi * Nd + t * 16 + mrow] = v;
        if (outB) outB[gi * obStride + obOff + t * 16 + mrow] = f2bf(v);
      }
    }
}

// ---------------- fused layer_out: lo = leaky(cat[cil] @ Wo1 + bo1) @ Wo2 + bo2, scatter-add ----
__global__ __launch_bounds__(32) void k_out(const u16* cat, const int* cil, const int* cin,
                                            const u16* wf1, const float* bo1,
                                            const u16* wf2, const float* bo2, float* psum) {
  __shared__ float ts[16][68];
  int lane = threadIdx.x;
  int mrow = lane & 15, half = lane >> 4;
  int n0 = blockIdx.x * 16;
  int row = cil[n0 + mrow];
  f32x8 z8 = {0.f,0.f,0.f,0.f,0.f,0.f,0.f,0.f};
  f32x8 acc[4]; acc[0]=z8; acc[1]=z8; acc[2]=z8; acc[3]=z8;
  #pragma unroll
  for (int kc = 0; kc < 4; ++kc) {
    bf16x16 a = load_a(cat, row * 128 + kc * 32 + half * 8);
    #pragma unroll
    for (int t = 0; t < 4; ++t) {
      bf16x16 b = load_b(wf1, kc, 4, t, lane);
      acc[t] = wmma_bf16(a, b, acc[t]);
    }
  }
  #pragma unroll
  for (int t = 0; t < 4; ++t)
    #pragma unroll
    for (int r = 0; r < 8; ++r) {
      int m = half * 8 + r;
      ts[m][t * 16 + mrow] = leakyf(acc[t][r] + bo1[t * 16 + mrow]);
    }
  __syncthreads();
  f32x8 a2[4]; a2[0]=z8; a2[1]=z8; a2[2]=z8; a2[3]=z8;
  #pragma unroll
  for (int kc = 0; kc < 2; ++kc) {
    BFrag f;
    #pragma unroll
    for (int e = 0; e < 16; ++e) {
      int kkv = kc * 32 + (e & 7) + ((e >> 3) << 4) + half * 8;
      f.h[e] = f2bf(ts[mrow][kkv]);
    }
    #pragma unroll
    for (int t = 0; t < 4; ++t) {
      bf16x16 b = load_b(wf2, kc, 4, t, lane);
      a2[t] = wmma_bf16(f.v, b, a2[t]);
    }
  }
  #pragma unroll
  for (int t = 0; t < 4; ++t)
    #pragma unroll
    for (int r = 0; r < 8; ++r) {
      int m = half * 8 + r;
      int seg = cin[n0 + m];
      atomicAdd(&psum[seg * 64 + t * 16 + mrow], a2[t][r] + bo2[t * 16 + mrow]);
    }
}

// ---------------- BN statistics / apply ----------------
__global__ __launch_bounds__(64) void k_colstats(const float* x, float* stats, int N, int C) {
  int j = threadIdx.x;
  int rows = (N + gridDim.x - 1) / gridDim.x;
  int r0 = blockIdx.x * rows, r1 = r0 + rows; if (r1 > N) r1 = N;
  float s = 0.f, ss = 0.f;
  for (int r = r0; r < r1; ++r) { float v = x[r * C + j]; s += v; ss += v * v; }
  atomicAdd(&stats[j], s); atomicAdd(&stats[C + j], ss);
}
// out_bf16 = (act?)((x-mu)*rs*g + b)    (act=1 -> leaky)
__global__ __launch_bounds__(256) void k_bn_act(const float* x, u16* outB, const float* g,
                                                const float* b, const float* stats,
                                                float invN, int n, int cmask, int C, int act) {
  int idx = blockIdx.x * 256 + threadIdx.x;
  if (idx >= n) return;
  int j = idx & cmask;
  float mu = stats[j] * invN;
  float var = stats[C + j] * invN - mu * mu;
  float rs = rsqrtf(var + 1e-5f);
  float v = (x[idx] - mu) * rs * g[j] + b[j];
  if (act) v = leakyf(v);
  outB[idx] = f2bf(v);
}
// out = leaky(bn(x) + res)  -> f32 (+opt bf16)
__global__ __launch_bounds__(256) void k_bn_res(const float* x, const float* res, float* outF,
                                                u16* outB, const float* g, const float* b,
                                                const float* stats, float invN, int n,
                                                int cmask, int C) {
  int idx = blockIdx.x * 256 + threadIdx.x;
  if (idx >= n) return;
  int j = idx & cmask;
  float mu = stats[j] * invN;
  float var = stats[C + j] * invN - mu * mu;
  float rs = rsqrtf(var + 1e-5f);
  float v = leakyf((x[idx] - mu) * rs * g[j] + b[j] + res[idx]);
  outF[idx] = v;
  if (outB) outB[idx] = f2bf(v);
}

// ---------------- logits + BCE loss ----------------
__device__ __forceinline__ int flat_of(const int* coors, int i) {
  int c0 = coors[i * 4 + 0], c1 = coors[i * 4 + 1], c2 = coors[i * 4 + 2], c3 = coors[i * 4 + 3];
  return ((c0 * SXC + c1) * SYC + c2) * SZC + c3;
}
__global__ __launch_bounds__(128) void k_logit(const float* vp, const int* nbrp, const float* Wlog,
                                               const float* blog, const int* coors,
                                               float* logit, u32* aux, int n) {
  int i = blockIdx.x * 128 + threadIdx.x;
  if (i >= n) return;
  float acc = blog[0];
  for (int k = 0; k < KK; ++k) {
    const float* vr = vp + nbrp[i * KK + k] * 64;
    const float* wr = Wlog + k * 64;
    #pragma unroll 8
    for (int c = 0; c < 64; ++c) acc += vr[c] * wr[c];
  }
  logit[i] = acc;
  atomicMax(&aux[flat_of(coors, i)], (u32)(i + 1));
}
__global__ __launch_bounds__(256) void k_scatterx(const float* logit, const int* coors,
                                                  const u32* aux, float* xd, int n) {
  int i = blockIdx.x * 256 + threadIdx.x;
  if (i >= n) return;
  int f = flat_of(coors, i);
  if (aux[f] == (u32)(i + 1)) xd[f] = logit[i];
}
__global__ __launch_bounds__(256) void k_loss(const float* xd, float* acc, int n) {
  __shared__ float red[256];
  float s = 0.f;
  for (int i = blockIdx.x * 256 + threadIdx.x; i < n; i += gridDim.x * 256) {
    float x = xd[i];
    s += fmaxf(x, 0.f) - x + log1pf(expf(-fabsf(x)));
  }
  red[threadIdx.x] = s; __syncthreads();
  for (int w = 128; w > 0; w >>= 1) { if (threadIdx.x < w) red[threadIdx.x] += red[threadIdx.x + w]; __syncthreads(); }
  if (threadIdx.x == 0) atomicAdd(acc, red[0]);
}

// ---------------- scatters / gathers ----------------
__global__ __launch_bounds__(256) void k_fs_scatter(const float* feats, const float* v,
                                                    const int* dsinv, float* dsum, float* dcnt,
                                                    u16* fsb, int n) {
  int idx = blockIdx.x * 256 + threadIdx.x;
  if (idx >= n) return;
  int nn = idx >> 6, j = idx & 63;
  float val = feats[idx] + v[idx];
  fsb[idx] = f2bf(val);
  int s = dsinv[nn];
  atomicAdd(&dsum[s * 64 + j], val);
  if (j == 0) atomicAdd(&dcnt[s], 1.f);
}
__global__ __launch_bounds__(256) void k_dsdiv(const float* dsum, const float* dcnt, u16* dsb, int n) {
  int idx = blockIdx.x * 256 + threadIdx.x;
  if (idx >= n) return;
  dsb[idx] = f2bf(dsum[idx] / fmaxf(dcnt[idx >> 6], 1.f));
}
__global__ __launch_bounds__(256) void k_cath(const u16* p3b, const int* dsinv, u16* cat, int n) {
  int idx = blockIdx.x * 256 + threadIdx.x;
  if (idx >= n) return;
  int nn = idx >> 6, j = idx & 63;
  cat[nn * 128 + 64 + j] = p3b[dsinv[nn] * 64 + j];
}
__global__ __launch_bounds__(256) void k_pcnt(const int* cin, float* pcnt, int n) {
  int i = blockIdx.x * 256 + threadIdx.x;
  if (i >= n) return;
  atomicAdd(&pcnt[cin[i]], 1.f);
}
__global__ __launch_bounds__(256) void k_final(const float* psum, const float* pcnt,
                                               const int* cin, float* out, int n) {
  int idx = blockIdx.x * 256 + threadIdx.x;
  if (idx >= n) return;
  int p = idx >> 6, j = idx & 63;
  int s = cin[p];
  out[idx] = psum[s * 64 + j] / fmaxf(pcnt[s], 1.f);
}
__global__ void k_losswrite(const float* acc, float* out) {
  out[0] = acc[0] * (1.f / (float)NCELL);
}

// ---------------- workspace layout (bytes) ----------------
static constexpr size_t OFF_WF11 = 0;
static constexpr size_t OFF_WF12 = 221184;
static constexpr size_t OFF_WF21 = 442368;
static constexpr size_t OFF_WF22 = 663552;
static constexpr size_t OFF_WFI  = 884736;
static constexpr size_t OFF_WFO1 = 892928;
static constexpr size_t OFF_WFO2 = 909312;
static constexpr size_t OFF_WFP1 = 917504;
static constexpr size_t OFF_WFP2 = 921600;
static constexpr size_t OFF_WFP3 = 923648;
static constexpr size_t OFF_STATS = 1048576;
static constexpr size_t OFF_LOGIT = 1114112;
static constexpr size_t OFF_A = 2097152;                 // 64 MB
static constexpr size_t OFF_B = OFF_A + 64000000;        // 64 MB
static constexpr size_t OFF_C = OFF_B + 64000000;        // 32 MB
static constexpr size_t OFF_D = OFF_C + 32000000;        // 32 MB
static constexpr size_t OFF_E = OFF_D + 32000000;        // 26 MB
static constexpr size_t WS_NEEDED = OFF_E + 26000000;

static inline int cdiv(int a, int b) { return (a + b - 1) / b; }

extern "C" void kernel_launch(void* const* d_in, const int* in_sizes, int n_in,
                              void* d_out, int out_size, void* d_ws, size_t ws_size,
                              hipStream_t stream) {
  if (ws_size < WS_NEEDED) return;
  char* ws = (char*)d_ws;

  // inputs (setup_inputs dict order)
  const float* feats  = (const float*)d_in[0];
  const float* featsp = (const float*)d_in[1];
  const float* Wv[4]  = {(const float*)d_in[2], (const float*)d_in[3], (const float*)d_in[4], (const float*)d_in[5]};
  const float* gB[4]  = {(const float*)d_in[6], (const float*)d_in[7], (const float*)d_in[8], (const float*)d_in[9]};
  const float* bB[4]  = {(const float*)d_in[10], (const float*)d_in[11], (const float*)d_in[12], (const float*)d_in[13]};
  const float* Wlog = (const float*)d_in[14]; const float* blog = (const float*)d_in[15];
  const float* Wi  = (const float*)d_in[16]; const float* bi  = (const float*)d_in[17];
  const float* Wp1 = (const float*)d_in[18]; const float* bp1 = (const float*)d_in[19];
  const float* gp1 = (const float*)d_in[20]; const float* bp1n= (const float*)d_in[21];
  const float* Wp2 = (const float*)d_in[22]; const float* bp2 = (const float*)d_in[23];
  const float* gp2 = (const float*)d_in[24]; const float* bp2n= (const float*)d_in[25];
  const float* Wp3 = (const float*)d_in[26]; const float* bp3 = (const float*)d_in[27];
  const float* Wo1 = (const float*)d_in[28]; const float* bo1 = (const float*)d_in[29];
  const float* Wo2 = (const float*)d_in[30]; const float* bo2 = (const float*)d_in[31];
  const int* nbr  = (const int*)d_in[32];
  const int* nbrp = (const int*)d_in[33];
  const int* coors= (const int*)d_in[34];
  const int* cil  = (const int*)d_in[35];
  const int* cin  = (const int*)d_in[36];
  const int* dsinv= (const int*)d_in[37];

  // workspace views
  u16* wfW[4] = {(u16*)(ws+OFF_WF11), (u16*)(ws+OFF_WF12), (u16*)(ws+OFF_WF21), (u16*)(ws+OFF_WF22)};
  u16* wfI  = (u16*)(ws+OFF_WFI);
  u16* wfO1 = (u16*)(ws+OFF_WFO1);
  u16* wfO2 = (u16*)(ws+OFF_WFO2);
  u16* wfP1 = (u16*)(ws+OFF_WFP1);
  u16* wfP2 = (u16*)(ws+OFF_WFP2);
  u16* wfP3 = (u16*)(ws+OFF_WFP3);
  float* stats = (float*)(ws+OFF_STATS);
  float* lossacc = stats + 12*128;
  float* logit = (float*)(ws+OFF_LOGIT);

  float* cbuf  = (float*)(ws+OFF_A);               // full conv out; later: partial conv out
  float* vp    = (float*)(ws+OFF_A+32000000);      // partial x1 -> vp (in place)
  u16*   cat   = (u16*)(ws+OFF_A);                 // after loss phase
  float* xbuf  = (float*)(ws+OFF_B);               // full x1 -> v (in place)
  u16*   hb    = (u16*)(ws+OFF_C);                 // full bn output (conv input)
  u16*   hbp   = (u16*)(ws+OFF_C);                 // partial bn output
  u16*   xbp   = (u16*)(ws+OFF_C+16000000);        // partial bf16 residual input
  u16*   fsb   = (u16*)(ws+OFF_C);                 // feat_sum bf16 (dense phase)
  u16*   xb    = (u16*)(ws+OFF_D);                 // full bf16 conv input
  u16*   dsb   = (u16*)(ws+OFF_D);                 // ds bf16 (dense phase)
  float* p1f   = (float*)(ws+OFF_D+12800000);      // p1/p2 f32
  u16*   p1b   = (u16*)(ws+OFF_D+25600000);        // p1b/p2b bf16
  u16*   p3b   = (u16*)(ws+OFF_D+12800000);        // p3 bf16 (after p2f dead)
  float* xd    = (float*)(ws+OFF_E);
  u32*   aux   = (u32*)(ws+OFF_E+5529600);
  float* dsum  = (float*)(ws+OFF_E);
  float* dcnt  = (float*)(ws+OFF_E+25600000);
  float* psum  = (float*)(ws+OFF_E);
  float* pcnt  = (float*)(ws+OFF_E+25600000);
  float* out   = (float*)d_out;

  // ---- phase 0: stats zero + weight swizzle ----
  k_zero<<<cdiv(12*128+16,256),256,0,stream>>>(stats, 12*128+16);
  for (int w = 0; w < 4; ++w)
    k_swz<<<cdiv(110592,256),256,0,stream>>>(Wv[w], wfW[w], KK*64, 64, 110592);
  k_swz<<<cdiv(4096,256),256,0,stream>>>(Wi,  wfI,  64, 64, 4096);
  k_swz<<<cdiv(8192,256),256,0,stream>>>(Wo1, wfO1, 128,64, 8192);
  k_swz<<<cdiv(4096,256),256,0,stream>>>(Wo2, wfO2, 64, 64, 4096);
  k_swz<<<cdiv(2048,256),256,0,stream>>>(Wp1, wfP1, 64, 32, 2048);
  k_swz<<<cdiv(1024,256),256,0,stream>>>(Wp2, wfP2, 32, 32, 1024);
  k_swz<<<cdiv(2048,256),256,0,stream>>>(Wp3, wfP3, 32, 64, 2048);

  const int TVOX = NVOX/16, TPART = cdiv(NPART,16), TPTS = NPTS/16, TDS = NDS/16;
  const int EVOX = NVOX*64, EPART = NPART*64;
  const float iVOX = 1.f/NVOX, iPART = 1.f/NPART, iDS = 1.f/NDS;

  // ---- full tensor: 2 SparseBasicBlocks ----
  k_cvt_bf16<<<cdiv(EVOX,256),256,0,stream>>>(feats, xb, EVOX);
  k_gconv<<<TVOX,32,0,stream>>>(xb, nbr, wfW[0], cbuf, NVOX);
  k_colstats<<<512,64,0,stream>>>(cbuf, stats+0*128, NVOX, 64);
  k_bn_act<<<cdiv(EVOX,256),256,0,stream>>>(cbuf, hb, gB[0], bB[0], stats+0*128, iVOX, EVOX, 63, 64, 1);
  k_gconv<<<TVOX,32,0,stream>>>(hb, nbr, wfW[1], cbuf, NVOX);
  k_colstats<<<512,64,0,stream>>>(cbuf, stats+1*128, NVOX, 64);
  k_bn_res<<<cdiv(EVOX,256),256,0,stream>>>(cbuf, feats, xbuf, xb, gB[1], bB[1], stats+1*128, iVOX, EVOX, 63, 64);
  k_gconv<<<TVOX,32,0,stream>>>(xb, nbr, wfW[2], cbuf, NVOX);
  k_colstats<<<512,64,0,stream>>>(cbuf, stats+2*128, NVOX, 64);
  k_bn_act<<<cdiv(EVOX,256),256,0,stream>>>(cbuf, hb, gB[2], bB[2], stats+2*128, iVOX, EVOX, 63, 64, 1);
  k_gconv<<<TVOX,32,0,stream>>>(hb, nbr, wfW[3], cbuf, NVOX);
  k_colstats<<<512,64,0,stream>>>(cbuf, stats+3*128, NVOX, 64);
  k_bn_res<<<cdiv(EVOX,256),256,0,stream>>>(cbuf, xbuf, xbuf, (u16*)nullptr, gB[3], bB[3], stats+3*128, iVOX, EVOX, 63, 64);
  // xbuf now holds v

  // ---- partial tensor: same blocks (shared weights) ----
  k_cvt_bf16<<<cdiv(EPART,256),256,0,stream>>>(featsp, xbp, EPART);
  k_gconv<<<TPART,32,0,stream>>>(xbp, nbrp, wfW[0], cbuf, NPART);
  k_colstats<<<512,64,0,stream>>>(cbuf, stats+4*128, NPART, 64);
  k_bn_act<<<cdiv(EPART,256),256,0,stream>>>(cbuf, hbp, gB[0], bB[0], stats+4*128, iPART, EPART, 63, 64, 1);
  k_gconv<<<TPART,32,0,stream>>>(hbp, nbrp, wfW[1], cbuf, NPART);
  k_colstats<<<512,64,0,stream>>>(cbuf, stats+5*128, NPART, 64);
  k_bn_res<<<cdiv(EPART,256),256,0,stream>>>(cbuf, featsp, vp, xbp, gB[1], bB[1], stats+5*128, iPART, EPART, 63, 64);
  k_gconv<<<TPART,32,0,stream>>>(xbp, nbrp, wfW[2], cbuf, NPART);
  k_colstats<<<512,64,0,stream>>>(cbuf, stats+6*128, NPART, 64);
  k_bn_act<<<cdiv(EPART,256),256,0,stream>>>(cbuf, hbp, gB[2], bB[2], stats+6*128, iPART, EPART, 63, 64, 1);
  k_gconv<<<TPART,32,0,stream>>>(hbp, nbrp, wfW[3], cbuf, NPART);
  k_colstats<<<512,64,0,stream>>>(cbuf, stats+7*128, NPART, 64);
  k_bn_res<<<cdiv(EPART,256),256,0,stream>>>(cbuf, vp, vp, (u16*)nullptr, gB[3], bB[3], stats+7*128, iPART, EPART, 63, 64);
  // vp now holds partial v

  // ---- mae logits + dense BCE loss ----
  k_zero<<<cdiv(NCELL,256),256,0,stream>>>(xd, NCELL);
  k_zero<<<cdiv(NCELL,256),256,0,stream>>>((float*)aux, NCELL);
  k_logit<<<cdiv(NPART,128),128,0,stream>>>(vp, nbrp, Wlog, blog, coors, logit, aux, NPART);
  k_scatterx<<<cdiv(NPART,256),256,0,stream>>>(logit, coors, aux, xd, NPART);
  k_loss<<<1024,256,0,stream>>>(xd, lossacc, NCELL);

  // ---- point encoder ----
  k_zero<<<cdiv(NDS*64,256),256,0,stream>>>(dsum, NDS*64);
  k_zero<<<cdiv(NDS,256),256,0,stream>>>(dcnt, NDS);
  k_fs_scatter<<<cdiv(EVOX,256),256,0,stream>>>(feats, xbuf, dsinv, dsum, dcnt, fsb, EVOX);
  k_dsdiv<<<cdiv(NDS*64,256),256,0,stream>>>(dsum, dcnt, dsb, NDS*64);
  // identity = leaky(feat_sum @ Wi + bi) -> cat[:, 0:64]
  k_dgemm<<<TVOX,32,0,stream>>>(fsb, (const int*)nullptr, wfI, bi, (float*)nullptr, cat, 128, 0, NVOX, 64, 64, 1);
  // PPmodel
  k_dgemm<<<TDS,32,0,stream>>>(dsb, (const int*)nullptr, wfP1, bp1, p1f, (u16*)nullptr, 0, 0, NDS, 64, 32, 1);
  k_colstats<<<512,32,0,stream>>>(p1f, stats+8*128, NDS, 32);
  k_bn_act<<<cdiv(NDS*32,256),256,0,stream>>>(p1f, p1b, gp1, bp1n, stats+8*128, iDS, NDS*32, 31, 32, 0);
  k_dgemm<<<TDS,32,0,stream>>>(p1b, (const int*)nullptr, wfP2, bp2, p1f, (u16*)nullptr, 0, 0, NDS, 32, 32, 1);
  k_colstats<<<512,32,0,stream>>>(p1f, stats+9*128, NDS, 32);
  k_bn_act<<<cdiv(NDS*32,256),256,0,stream>>>(p1f, p1b, gp2, bp2n, stats+9*128, iDS, NDS*32, 31, 32, 0);
  k_dgemm<<<TDS,32,0,stream>>>(p1b, (const int*)nullptr, wfP3, bp3, (float*)nullptr, p3b, 64, 0, NDS, 32, 64, 1);
  k_cath<<<cdiv(EVOX,256),256,0,stream>>>(p3b, dsinv, cat, EVOX);
  // layer_out fused + scatter_mean
  k_zero<<<cdiv(NNEXT*64,256),256,0,stream>>>(psum, NNEXT*64);
  k_zero<<<cdiv(NNEXT,256),256,0,stream>>>(pcnt, NNEXT);
  k_pcnt<<<cdiv(NPTS,256),256,0,stream>>>(cin, pcnt, NPTS);
  k_out<<<TPTS,32,0,stream>>>(cat, cil, cin, wfO1, bo1, wfO2, bo2, psum);
  k_final<<<cdiv(OUTFEA,256),256,0,stream>>>(psum, pcnt, cin, out, OUTFEA);
  k_losswrite<<<1,1,0,stream>>>(lossacc, out + OUTFEA);
}
